// EdgeConvLayer_5042291606133
// MI455X (gfx1250) — compile-verified, matrix operand order checked by reading, not run on hardware
//
#include <hip/hip_runtime.h>
#include <hip/hip_bf16.h>

// ---------------------------------------------------------------------------
// EdgeConv: msg = [x_i, x_j - x_i] @ W^T + b ; out = segment_max(msg, dst)
// Factored: msg = A[i] + B[j] + b  with A = x@(W1-W2)^T, B = x@W2^T
//           out[i] = A[i] + b + max_{j in N(i)} B[j]
// GEMMs via v_wmma_f32_16x16x32_bf16; max via global_atomic_max_u32 key trick.
// ---------------------------------------------------------------------------

typedef __attribute__((ext_vector_type(8)))  __bf16 v8bf;
typedef __attribute__((ext_vector_type(16))) __bf16 v16bf;
typedef __attribute__((ext_vector_type(8)))  float  v8f;

#define CH 128
#define NODES_PER_TILE 16

__device__ __forceinline__ unsigned short f32_to_bf16_rne(float f) {
    unsigned u = __float_as_uint(f);
    unsigned r = u + 0x7FFFu + ((u >> 16) & 1u);   // round-to-nearest-even
    return (unsigned short)(r >> 16);
}

// monotonic float -> uint mapping (order-preserving): fmax == umax on keys
__device__ __forceinline__ unsigned enc_key(float f) {
    unsigned u = __float_as_uint(f);
    return (u & 0x80000000u) ? ~u : (u | 0x80000000u);
}
__device__ __forceinline__ float dec_key(unsigned k) {
    unsigned u = (k & 0x80000000u) ? (k & 0x7FFFFFFFu) : ~k;
    return __uint_as_float(u);
}

__device__ __forceinline__ v16bf load_frag(const __bf16* p) {
    // 16 bf16 = two contiguous 16-byte chunks 16 bf16 apart in K
    v8bf lo = *(const v8bf*)(p);
    v8bf hi = *(const v8bf*)(p + 16);
    return __builtin_shufflevector(lo, hi,
        0, 1, 2, 3, 4, 5, 6, 7, 8, 9, 10, 11, 12, 13, 14, 15);
}

// ---- convert x (fp32) -> bf16 ----------------------------------------------
__global__ void conv_x_kernel(const float* __restrict__ x,
                              unsigned short* __restrict__ xb, int n_elem) {
    int i = blockIdx.x * blockDim.x + threadIdx.x;
    if (i < n_elem) xb[i] = f32_to_bf16_rne(x[i]);
}

// ---- build Wd = W1 - W2 and W2 (bf16), each [128][128] row-major ------------
__global__ void conv_w_kernel(const float* __restrict__ W,   // [128][256]
                              unsigned short* __restrict__ wd,
                              unsigned short* __restrict__ w2) {
    int i = blockIdx.x * blockDim.x + threadIdx.x;    // 128*128 elements
    if (i >= CH * CH) return;
    int n = i >> 7, k = i & 127;
    float w1v = W[n * (2 * CH) + k];
    float w2v = W[n * (2 * CH) + CH + k];
    wd[i] = f32_to_bf16_rne(w1v - w2v);
    w2[i] = f32_to_bf16_rne(w2v);
}

// ---- GEMM: out[N][128] = x[N][128] @ Wt^T, Wt row-major [128][128] bf16 -----
// One wave per 16-row M tile; full 128-col output tile in 8 v8f accumulators.
// All 8 B fragments of a K-step loaded into distinct regs before the WMMA
// burst so loads pipeline with matrix issue (partial s_wait_loadcnt).
__global__ void __launch_bounds__(256)
gemm_wmma_kernel(const unsigned short* __restrict__ xb,
                 const unsigned short* __restrict__ wt,
                 float* __restrict__ out, int n_mtiles) {
    int wave = (blockIdx.x * blockDim.x + threadIdx.x) >> 5;
    if (wave >= n_mtiles) return;
    int lane = threadIdx.x & 31;
    int lo   = lane & 15;      // row (A) / col (B) within tile
    int half = lane >> 4;      // K sub-block select per ISA VGPR layout

    v8f acc[8];
#pragma unroll
    for (int i = 0; i < 8; ++i) acc[i] = 0;

    const __bf16* arow  = (const __bf16*)xb + (wave * NODES_PER_TILE + lo) * CH;
    const __bf16* wbase = (const __bf16*)wt + lo * CH;   // row 'lo' of W tile 0

#pragma unroll
    for (int kb = 0; kb < CH; kb += 32) {
        v16bf afrag = load_frag(arow + kb + half * 8);
        v16bf bfr[8];
#pragma unroll
        for (int nt = 0; nt < 8; ++nt)
            bfr[nt] = load_frag(wbase + nt * 16 * CH + kb + half * 8);
#pragma unroll
        for (int nt = 0; nt < 8; ++nt)
            acc[nt] = __builtin_amdgcn_wmma_f32_16x16x32_bf16(
                false, afrag, false, bfr[nt], (short)0, acc[nt], false, false);
    }

    // C/D layout: VGPR r -> M = r + 8*half, N = lo
    int row0 = wave * NODES_PER_TILE + half * 8;
#pragma unroll
    for (int nt = 0; nt < 8; ++nt)
#pragma unroll
        for (int r = 0; r < 8; ++r)
            out[(row0 + r) * CH + nt * 16 + lo] = acc[nt][r];
}

// ---- init keys (stored in d_out as uint) to 0 (== "-inf" sentinel) ----------
__global__ void init_keys_kernel(unsigned* __restrict__ keys, int n_elem) {
    int i = blockIdx.x * blockDim.x + threadIdx.x;
    if (i < n_elem) keys[i] = 0u;
}

// ---- per-edge scatter-max: one wave per edge, 4 channels per lane -----------
__global__ void __launch_bounds__(256)
scatter_max_kernel(const long long* __restrict__ ei,   // [2][E] int64
                   const float* __restrict__ Bn,       // [N][128]
                   unsigned* __restrict__ keys,        // [N][128]
                   int E) {
    int t = blockIdx.x * blockDim.x + threadIdx.x;
    int edge = t >> 5;
    if (edge >= E) return;
    int lane = t & 31;
    int src = (int)ei[edge];
    int dst = (int)ei[E + edge];
    const float4 v = *(const float4*)(Bn + src * CH + lane * 4);
    unsigned* kp = keys + dst * CH + lane * 4;
    atomicMax(kp + 0, enc_key(v.x));
    atomicMax(kp + 1, enc_key(v.y));
    atomicMax(kp + 2, enc_key(v.z));
    atomicMax(kp + 3, enc_key(v.w));
}

// ---- finalize in place: out = key ? (A + bias + dec(key)) : 0 ---------------
__global__ void finalize_kernel(unsigned* __restrict__ keys_and_out,
                                const float* __restrict__ A,
                                const float* __restrict__ bias, int n_elem) {
    int i = blockIdx.x * blockDim.x + threadIdx.x;
    if (i >= n_elem) return;
    unsigned k = keys_and_out[i];
    float res = 0.0f;
    if (k != 0u) {
        float v = A[i] + bias[i & (CH - 1)] + dec_key(k);
        res = __builtin_isfinite(v) ? v : 0.0f;
    }
    ((float*)keys_and_out)[i] = res;
}

extern "C" void kernel_launch(void* const* d_in, const int* in_sizes, int n_in,
                              void* d_out, int out_size, void* d_ws, size_t ws_size,
                              hipStream_t stream) {
    const float*     x  = (const float*)d_in[0];      // [N][128]
    const float*     W  = (const float*)d_in[1];      // [128][256]
    const float*     b  = (const float*)d_in[2];      // [128]
    const long long* ei = (const long long*)d_in[3];  // [2][E] int64

    const int N = in_sizes[0] / CH;          // 10000
    const int E = in_sizes[3] / 2;           // 640000
    const int n_mtiles = N / NODES_PER_TILE; // 625 (N divisible by 16)

    // workspace layout (16B-aligned): xbf | Wd | W2 | A | B  (~12.9 MB)
    char* ws = (char*)d_ws;
    unsigned short* xbf  = (unsigned short*)(ws);
    size_t off = (size_t)N * CH * 2;
    unsigned short* wdbf = (unsigned short*)(ws + off); off += CH * CH * 2;
    unsigned short* w2bf = (unsigned short*)(ws + off); off += CH * CH * 2;
    float* Aout = (float*)(ws + off); off += (size_t)N * CH * 4;
    float* Bout = (float*)(ws + off);

    unsigned* keys = (unsigned*)d_out;   // keys live in d_out, decoded in place

    const int nex = N * CH;
    conv_x_kernel<<<(nex + 255) / 256, 256, 0, stream>>>(x, xbf, nex);
    conv_w_kernel<<<(CH * CH + 255) / 256, 256, 0, stream>>>(W, wdbf, w2bf);

    const int gemm_blocks = (n_mtiles * 32 + 255) / 256;
    gemm_wmma_kernel<<<gemm_blocks, 256, 0, stream>>>(xbf, wdbf, Aout, n_mtiles);
    gemm_wmma_kernel<<<gemm_blocks, 256, 0, stream>>>(xbf, w2bf, Bout, n_mtiles);

    init_keys_kernel<<<(nex + 255) / 256, 256, 0, stream>>>(keys, nex);
    scatter_max_kernel<<<(E * 32 + 255) / 256, 256, 0, stream>>>(ei, Bout, keys, E);
    finalize_kernel<<<(nex + 255) / 256, 256, 0, stream>>>(keys, Aout, b, nex);
}
